// RetBlock_4200478015632
// MI455X (gfx1250) — matrix operand order repaired and checked by reading, compiled, not run on hardware
//
#include <hip/hip_runtime.h>
#include <stdint.h>

// ---------------------------------------------------------------------------
// MI455X (gfx1250) retention block:
//   kernel 0a: x  (fp32 -> bf16)                 streaming convert
//   kernel 0b: wT (768 x 2048 bf16)              transposed fused weights
//   kernel 1 : QKV projection GEMM (wmma bf16) + xpos epilogue
//              Q,K stored row-major bf16; V stored transposed (B,Dh,S) bf16
//   kernel 2 : fused causal retention: (Q K^T * gamma^(s-t)) V, wmma bf16,
//              fp32 accum. Q panel staged into LDS via
//              global_load_async_to_lds_b128 (ASYNCcnt), S-tile bounced
//              through LDS for the AV A-operand, ds_add_f32 reduction.
// ---------------------------------------------------------------------------

typedef __attribute__((ext_vector_type(8)))  float  v8f;
typedef __attribute__((ext_vector_type(16))) __bf16 v16bf;
typedef __attribute__((ext_vector_type(8)))  __bf16 v8bf;

#define LN_GAMMA   (-0.031748698f)   // ln(0.96875)
#define LN1E4_D128 (0.07195578f)     // ln(10000)/128

static constexpr int Bsz = 4, Seq = 4096, Hdim = 2048, Dh = 256;
static constexpr int Mrows = Bsz * Seq;        // 16384
static constexpr int NCols = 3 * Dh;           // 768

__device__ __forceinline__ v8f wmma_bf16(v16bf a, v16bf b, v8f c) {
  return __builtin_amdgcn_wmma_f32_16x16x32_bf16(
      /*neg_a=*/false, a, /*neg_b=*/false, b,
      /*c_mod=*/(short)0, c, /*reuse_a=*/false, /*reuse_b=*/false);
}

__device__ __forceinline__ v16bf cat8(v8bf lo, v8bf hi) {
  return __builtin_shufflevector(lo, hi, 0, 1, 2, 3, 4, 5, 6, 7,
                                         8, 9, 10, 11, 12, 13, 14, 15);
}

// ---------------------------- kernel 0a: x -> bf16 -------------------------
__global__ void convert_x_kernel(const float* __restrict__ x,
                                 __bf16* __restrict__ xb, int n) {
  for (int i = blockIdx.x * blockDim.x + threadIdx.x; i < n;
       i += gridDim.x * blockDim.x)
    xb[i] = (__bf16)x[i];
}

// ------------------------- kernel 0b: fused wT (768,2048) ------------------
__global__ void build_wT_kernel(const float* __restrict__ wq,
                                const float* __restrict__ wk,
                                const float* __restrict__ wv,
                                __bf16* __restrict__ wT) {
  int idx = blockIdx.x * blockDim.x + threadIdx.x;  // over 768*2048
  if (idx >= NCols * Hdim) return;
  int oc = idx / Hdim, ic = idx - oc * Hdim;
  const float* w = (oc < Dh) ? wq : (oc < 2 * Dh) ? wk : wv;
  wT[idx] = (__bf16)w[ic * Dh + (oc & (Dh - 1))];
}

// ------------- kernel 1: QKV projection (wmma) + xpos epilogue -------------
// grid = (Mrows/16, 3), block = 128 (4 waves). Wave w owns 64 output cols.
__global__ __launch_bounds__(128) void qkv_xpos_kernel(
    const __bf16* __restrict__ xb,  // (16384, 2048)
    const __bf16* __restrict__ wT,  // (768, 2048)
    __bf16* __restrict__ Qb,        // (16384, 256)
    __bf16* __restrict__ Kb,        // (16384, 256)
    __bf16* __restrict__ Vt)        // (4, 256, 4096)
{
  const int lane = threadIdx.x & 31;
  const int wave = threadIdx.x >> 5;
  const int ln = lane & 15, hi = lane >> 4;
  const int m0 = blockIdx.x * 16;
  const int n0 = blockIdx.y * 256 + wave * 64;   // [0,768) in 64-col strips

  v8f acc[4];
#pragma unroll
  for (int i = 0; i < 4; ++i) acc[i] = (v8f){0, 0, 0, 0, 0, 0, 0, 0};

  const __bf16* arow = xb + (size_t)(m0 + ln) * Hdim;

  for (int k0 = 0; k0 < Hdim; k0 += 32) {
    v16bf A = *(const v16bf*)(arow + k0 + hi * 16);       // K 0-15 | 16-31
#pragma unroll
    for (int nt = 0; nt < 4; ++nt) {
      const __bf16* brow =
          wT + (size_t)(n0 + nt * 16 + ln) * Hdim + k0 + hi * 16;
      v16bf Bv = *(const v16bf*)brow;
      acc[nt] = wmma_bf16(A, Bv, acc[nt]);
    }
  }

  // ----- epilogue: xpos(q), xpos(k, downscale), passthrough v -------------
  const int which = n0 >> 8;  // 0=q 1=k 2=v (wave-uniform)
#pragma unroll
  for (int nt = 0; nt < 4; ++nt) {
#pragma unroll
    for (int r = 0; r < 8; ++r) {
      float val = acc[nt][r];
      int srow = m0 + r + 8 * hi;          // global row = b*4096 + s
      int s = srow & (Seq - 1);
      int n = n0 + nt * 16 + ln;
      int d = n & (Dh - 1);
      if (which == 2) {
        // V transposed: (b, d, s)
        Vt[((size_t)((srow >> 12) * Dh + d)) * Seq + s] = (__bf16)val;
      } else {
        float par = __shfl_xor(val, 1, 32);  // rotate-every-two partner
        int j = d >> 1;
        float scv = ((float)(2 * j) + 102.4f) * (1.0f / 358.4f);
        float p = ((float)s - 2048.0f) * (1.0f / 512.0f);
        if (which == 1) p = -p;  // downscale for K
        float scale = __expf(p * __logf(scv));
        float freq = __expf(-(float)j * LN1E4_D128);
        float sn, cs;
        __sincosf((float)s * freq, &sn, &cs);
        float cf = cs * scale, sf = sn * scale;
        float out = (d & 1) ? (val * cf + par * sf) : (val * cf - par * sf);
        __bf16* dst = (which == 0) ? Qb : Kb;
        dst[(size_t)srow * Dh + d] = (__bf16)out;
      }
    }
  }
}

// ------------- kernel 2: fused causal retention (flash style) --------------
// grid = (Seq/16, B), block = 128 (4 waves). Block owns 16 output rows x 256.
// Q row-panel (8 KB, contiguous) is block-staged into LDS with async copies;
// waves split the 32-wide t-tiles round-robin; partial outputs reduced via
// ds_add_f32 into LDS.
__global__ __launch_bounds__(128, 1) void retention_kernel(
    const __bf16* __restrict__ Qb, const __bf16* __restrict__ Kb,
    const __bf16* __restrict__ Vt, float* __restrict__ out)
{
  __shared__ float red[16 * 256];          // 16 KB fp32 reduction tile
  __shared__ __bf16 qtile[16 * 256];       //  8 KB Q row-panel (bf16)
  __shared__ __bf16 stile[4][16 * 32];     // per-wave decayed S-tile (bf16)

  const int lane = threadIdx.x & 31;
  const int wave = threadIdx.x >> 5;
  const int ln = lane & 15, hi = lane >> 4;
  const int s0 = blockIdx.x * 16;
  const int b = blockIdx.y;

  // ---- async block-copy of the contiguous 8 KB Q panel into LDS ----------
  {
    const char* src = (const char*)(Qb + (size_t)(b * Seq + s0) * Dh);
    const unsigned dst0 = (unsigned)(size_t)(&qtile[0]);
#pragma unroll
    for (int i = 0; i < 4; ++i) {
      const int c = (int)threadIdx.x + 128 * i;     // 512 x 16B chunks
      uint64_t ga = (uint64_t)(src + c * 16);
      unsigned ld = dst0 + c * 16;
      asm volatile("global_load_async_to_lds_b128 %0, %1, off"
                   :: "v"(ld), "v"(ga) : "memory");
    }
  }

  for (int i = threadIdx.x; i < 16 * 256; i += 128) red[i] = 0.f;

  asm volatile("s_wait_asynccnt 0" ::: "memory");
  __syncthreads();   // all waves' async chunks + red zeroing visible

  // Q A-operands for all 8 Dh slices (row = s0 + ln), read from LDS
  const __bf16* qr = &qtile[ln * 256];
  v16bf Aq[8];
#pragma unroll
  for (int kk = 0; kk < 8; ++kk) {
    Aq[kk] = *(const v16bf*)(qr + kk * 32 + hi * 16);
  }

  v8f acc[16];
#pragma unroll
  for (int i = 0; i < 16; ++i) acc[i] = (v8f){0, 0, 0, 0, 0, 0, 0, 0};

  const float gpow[8] = {1.0f, 0.96875f, 0.9384765625f, 0.909149169921875f,
                         0.880738258361816f, 0.853215187788f,
                         0.826552213170f, 0.800722456509f};

  const int nsteps = (s0 + 16 + 31) >> 5;                    // t tiles needed
  const int step_lo = (s0 > 3328) ? ((s0 - 3328) >> 5) : 0;  // gamma^d == 0
  bool did = false;

  const __bf16* kbat = Kb + (size_t)(b * Seq) * Dh;
  const __bf16* vbat = Vt + (size_t)(b * Dh) * Seq;
  __bf16* sw = &stile[wave][0];

  for (int st = step_lo + wave; st < nsteps; st += 4) {
    did = true;
    const int t0 = st << 5;

    if (st + 4 < nsteps) {  // token prefetch of next assigned tile
      __builtin_prefetch(kbat + (size_t)(t0 + 128) * Dh, 0, 0);
      __builtin_prefetch(vbat + t0 + 128, 0, 0);
    }

    // ---- S = Q K^T over Dh=256 (two 16x16 n-tiles: t_rel 0-15, 16-31) ----
    v8f sa0 = (v8f){0, 0, 0, 0, 0, 0, 0, 0};
    v8f sa1 = (v8f){0, 0, 0, 0, 0, 0, 0, 0};
    const __bf16* kb0 = kbat + (size_t)(t0 + ln) * Dh + hi * 16;
    const __bf16* kb1 = kbat + (size_t)(t0 + 16 + ln) * Dh + hi * 16;
#pragma unroll
    for (int kk = 0; kk < 8; ++kk)
      sa0 = wmma_bf16(Aq[kk], *(const v16bf*)(kb0 + kk * 32), sa0);
#pragma unroll
    for (int kk = 0; kk < 8; ++kk)
      sa1 = wmma_bf16(Aq[kk], *(const v16bf*)(kb1 + kk * 32), sa1);

    // ---- causal decay gamma^(s-t), write 16x32 bf16 S-tile to LDS --------
    const int srb = s0 + 8 * hi;
    const int tl0 = t0 + ln, tl1 = t0 + 16 + ln;
    const float base0 = __expf((float)(srb - tl0) * LN_GAMMA);
    const float base1 = __expf((float)(srb - tl1) * LN_GAMMA);
#pragma unroll
    for (int r = 0; r < 8; ++r) {
      float w0 = (srb + r - tl0 >= 0) ? base0 * gpow[r] : 0.f;
      float w1 = (srb + r - tl1 >= 0) ? base1 * gpow[r] : 0.f;
      sw[(r + 8 * hi) * 32 + ln] = (__bf16)(sa0[r] * w0);
      sw[(r + 8 * hi) * 32 + 16 + ln] = (__bf16)(sa1[r] * w1);
    }
    asm volatile("s_wait_dscnt 0" ::: "memory");

    // ---- reload as A-operand (16x32) -------------------------------------
    v16bf As = *(const v16bf*)(sw + ln * 32 + hi * 16);

    // ---- O += S * V  (V^T rows give contiguous K runs) -------------------
    const __bf16* vb = vbat + t0 + hi * 16;
#pragma unroll
    for (int nt = 0; nt < 16; ++nt) {
      v16bf Bv = *(const v16bf*)(vb + (size_t)(nt * 16 + ln) * Seq);
      acc[nt] = wmma_bf16(As, Bv, acc[nt]);
    }
  }

  // ---- cross-wave reduction via LDS float atomics ------------------------
  if (did) {
#pragma unroll
    for (int nt = 0; nt < 16; ++nt)
#pragma unroll
      for (int r = 0; r < 8; ++r)
        atomicAdd(&red[(r + 8 * hi) * 256 + nt * 16 + ln], acc[nt][r]);
  }
  __syncthreads();

  float* obase = out + (size_t)(b * Seq + s0) * Dh;
  for (int i = threadIdx.x; i < 16 * 256; i += 128) obase[i] = red[i];
}

// ---------------------------------------------------------------------------
extern "C" void kernel_launch(void* const* d_in, const int* in_sizes, int n_in,
                              void* d_out, int out_size, void* d_ws,
                              size_t ws_size, hipStream_t stream) {
  const float* x  = (const float*)d_in[0];
  const float* wq = (const float*)d_in[1];
  const float* wk = (const float*)d_in[2];
  const float* wv = (const float*)d_in[3];
  float* outp = (float*)d_out;

  // workspace layout (bf16)
  char* ws = (char*)d_ws;
  const size_t xb_bytes = (size_t)Mrows * Hdim * 2;   // 67.1 MB
  const size_t wT_bytes = (size_t)NCols * Hdim * 2;   //  3.1 MB
  const size_t qk_bytes = (size_t)Mrows * Dh * 2;     //  8.4 MB each
  __bf16* xb = (__bf16*)ws;
  __bf16* wT = (__bf16*)(ws + xb_bytes);
  __bf16* Qb = (__bf16*)(ws + xb_bytes + wT_bytes);
  __bf16* Kb = (__bf16*)(ws + xb_bytes + wT_bytes + qk_bytes);
  __bf16* Vt = (__bf16*)(ws + xb_bytes + wT_bytes + 2 * qk_bytes);
  if (ws_size < xb_bytes + wT_bytes + 3 * qk_bytes) return;

  const int nx = Mrows * Hdim;
  convert_x_kernel<<<4096, 256, 0, stream>>>(x, xb, nx);
  build_wT_kernel<<<(NCols * Hdim + 255) / 256, 256, 0, stream>>>(wq, wk, wv,
                                                                  wT);
  qkv_xpos_kernel<<<dim3(Mrows / 16, 3), 128, 0, stream>>>(xb, wT, Qb, Kb, Vt);
  retention_kernel<<<dim3(Seq / 16, Bsz), 128, 0, stream>>>(Qb, Kb, Vt, outp);
}
